// LocalCausalSelfAttention_50225347560052
// MI455X (gfx1250) — matrix operand order repaired and checked by reading, not compiled, not run on hardware
//
#include <hip/hip_runtime.h>
#include <cmath>

// ---------------------------------------------------------------- types
typedef __bf16 bf16;
typedef __attribute__((ext_vector_type(16))) __bf16 v16bf;
typedef __attribute__((ext_vector_type(2)))  __bf16 bf16x2;
typedef __attribute__((ext_vector_type(8)))  float  v8f;

// ---------------------------------------------------------------- problem constants
#define BATCH    2
#define T_SEQ    2048
#define CDIM     1024
#define N_HEADS  16
#define HEAD_DIM 64
#define WINDOW   128
#define M_ROWS   (BATCH * T_SEQ)          // 4096
#define LOG_ROPE_BASE 9.210340371976184f  // ln(10000)

// ---------------------------------------------------------------- ws layout (bytes)
static constexpr size_t OFF_XB     = 0;                                  // x bf16 [4096,1024]
static constexpr size_t OFF_WQKVT  = OFF_XB     + (size_t)M_ROWS*CDIM*2; // WqkvT bf16 [3072,1024]
static constexpr size_t OFF_WPROJT = OFF_WQKVT  + (size_t)3*CDIM*CDIM*2; // WprojT bf16 [1024,1024]
static constexpr size_t OFF_QKV    = OFF_WPROJT + (size_t)CDIM*CDIM*2;   // qkv f32 [4096,3072]
static constexpr size_t OFF_QB     = OFF_QKV    + (size_t)M_ROWS*3*CDIM*4; // q bf16 [B,H,T,64]
static constexpr size_t OFF_KB     = OFF_QB     + (size_t)M_ROWS*CDIM*2;   // k bf16 [B,H,T,64]
static constexpr size_t OFF_VT     = OFF_KB     + (size_t)M_ROWS*CDIM*2;   // v bf16 [B,H,64,T]
static constexpr size_t OFF_ATTNB  = OFF_VT     + (size_t)M_ROWS*CDIM*2;   // attn bf16 [4096,1024]

// ---------------------------------------------------------------- converts
__global__ void f32_to_bf16_kernel(const float* __restrict__ in, bf16* __restrict__ out, int n) {
    int i = blockIdx.x * blockDim.x + threadIdx.x;
    if (i < n) out[i] = (bf16)in[i];
}

// in: [K,N] f32 row-major  ->  out: [N,K] bf16 (transposed)
__global__ void transpose_to_bf16_kernel(const float* __restrict__ in, bf16* __restrict__ out,
                                         int K, int N) {
    int i = blockIdx.x * blockDim.x + threadIdx.x;
    if (i >= K * N) return;
    int n = i % N, k = i / N;
    out[(size_t)n * K + k] = (bf16)in[i];
}

// ---------------------------------------------------------------- LDS-blocked bf16 WMMA GEMM
// C[M,N] = A[M,K] * Bt[N,K]^T + bias[N].  Block = 256 threads = 8 waves.
// Block tile 128(M) x 128(N); wave tile 32 x 64 (waves arranged 4x2).
// K-slices of 32 are double-buffered in LDS via global_load_async_to_lds_b128 (ASYNCcnt).
__global__ __launch_bounds__(256) void gemm_bf16_kernel(
    const bf16* __restrict__ A, const bf16* __restrict__ Bt,
    const float* __restrict__ bias, float* __restrict__ C,
    int M, int N, int K)
{
    __shared__ __align__(16) bf16 ldsA[2][128 * 32];   // [buf][row][k]  8KB each
    __shared__ __align__(16) bf16 ldsB[2][128 * 32];

    const int tilesN = N >> 7;
    const int m0b = (int)(blockIdx.x / tilesN) * 128;
    const int n0b = (int)(blockIdx.x % tilesN) * 128;

    const int tid  = threadIdx.x;
    const int lane = tid & 31;
    const int half = lane >> 4;
    const int l16  = lane & 15;
    const int wave = tid >> 5;
    const int m0   = (wave >> 1) * 32;   // local M offset of this wave
    const int n0   = (wave & 1) * 64;    // local N offset of this wave

    const uintptr_t baseA = (uintptr_t)A;
    const uintptr_t baseB = (uintptr_t)Bt;

    v8f acc[2][4];
#pragma unroll
    for (int s = 0; s < 2; ++s)
#pragma unroll
        for (int nt = 0; nt < 4; ++nt)
#pragma unroll
            for (int r = 0; r < 8; ++r) acc[s][nt][r] = 0.f;

    // Stage one 128x32 A slice + 128x32 Bt slice into ldsA/ldsB[buf].
    // 512 16-byte chunks per matrix; 256 threads x 2 chunks; 4 async ops/thread.
    auto stage = [&](int buf, int kc) {
#pragma unroll
        for (int i = 0; i < 2; ++i) {
            int c   = tid * 2 + i;
            int row = c >> 2;            // 0..127
            int seg = (c & 3) * 8;       // element offset within the 32-wide K slice
            uint32_t ga = (uint32_t)(((m0b + row) * K + kc + seg) * 2);
            uint32_t la = (uint32_t)(uintptr_t)&ldsA[buf][row * 32 + seg];
            asm volatile("global_load_async_to_lds_b128 %0, %1, %2"
                         :: "v"(la), "v"(ga), "s"(baseA) : "memory");
            uint32_t gb = (uint32_t)(((n0b + row) * K + kc + seg) * 2);
            uint32_t lb = (uint32_t)(uintptr_t)&ldsB[buf][row * 32 + seg];
            asm volatile("global_load_async_to_lds_b128 %0, %1, %2"
                         :: "v"(lb), "v"(gb), "s"(baseB) : "memory");
        }
    };

    stage(0, 0);
    for (int kc = 0; kc < K; kc += 32) {
        int buf = (kc >> 5) & 1;
        if (kc + 32 < K) {
            stage(buf ^ 1, kc + 32);
            // 4 newest (next buffer) may remain outstanding; current buffer is done.
            asm volatile("s_wait_asynccnt 0x4" ::: "memory");
        } else {
            asm volatile("s_wait_asynccnt 0x0" ::: "memory");
        }
        __syncthreads();

        v16bf a[2];
#pragma unroll
        for (int s = 0; s < 2; ++s) {
            const bf16* arow = &ldsA[buf][(m0 + s * 16 + l16) * 32 + half * 16];
#pragma unroll
            for (int r = 0; r < 8; ++r) {
                int kb = 2 * r;
                bf16x2 p = *reinterpret_cast<const bf16x2*>(arow + kb);
                a[s][2 * r] = p[0]; a[s][2 * r + 1] = p[1];
            }
        }
#pragma unroll
        for (int nt = 0; nt < 4; ++nt) {
            const bf16* brow = &ldsB[buf][(n0 + nt * 16 + l16) * 32 + half * 16];
            v16bf b;
#pragma unroll
            for (int r = 0; r < 8; ++r) {
                bf16x2 p = *reinterpret_cast<const bf16x2*>(brow + 2 * r);
                b[2 * r] = p[0]; b[2 * r + 1] = p[1];
            }
            acc[0][nt] = __builtin_amdgcn_wmma_f32_16x16x32_bf16(false, a[0], false, b,
                                                                 (short)0, acc[0][nt], false, false);
            acc[1][nt] = __builtin_amdgcn_wmma_f32_16x16x32_bf16(false, a[1], false, b,
                                                                 (short)0, acc[1][nt], false, false);
        }
        __syncthreads();
    }

#pragma unroll
    for (int s = 0; s < 2; ++s)
#pragma unroll
        for (int nt = 0; nt < 4; ++nt) {
            int col  = n0b + n0 + nt * 16 + l16;
            float bv = bias[col];
#pragma unroll
            for (int r = 0; r < 8; ++r) {
                int mg = m0b + m0 + s * 16 + r + half * 8;
                C[(size_t)mg * N + col] = acc[s][nt][r] + bv;
            }
        }
}

// ---------------------------------------------------------------- RoPE + head split
// qkv f32 [B*T, 3C]  ->  q,k bf16 [B,H,T,64] (rotary applied), v bf16 [B,H,64,T] (transposed)
__global__ void rope_split_kernel(const float* __restrict__ qkv, bf16* __restrict__ qb,
                                  bf16* __restrict__ kb, bf16* __restrict__ vT)
{
    int i = blockIdx.x * blockDim.x + threadIdx.x;
    const int PAIRS = BATCH * T_SEQ * N_HEADS * (HEAD_DIM / 2);
    if (i >= PAIRS) return;
    int dp = i & 31;                 // pair index within head (HEAD_DIM/2 = 32)
    int h  = (i >> 5) & (N_HEADS - 1);
    int t  = (i >> 9) & (T_SEQ - 1);
    int b  = i >> 20;                // / (32*16*2048)
    size_t row = (size_t)b * T_SEQ + t;
    const float* base = qkv + row * (3 * CDIM) + h * HEAD_DIM + 2 * dp;
    float q0 = base[0],        q1 = base[1];
    float k0 = base[CDIM],     k1 = base[CDIM + 1];
    float v0 = base[2 * CDIM], v1 = base[2 * CDIM + 1];
    float inv = __expf(-((float)(2 * dp) / (float)HEAD_DIM) * LOG_ROPE_BASE);
    float ang = (float)t * inv;
    float s, c;
    sincosf(ang, &s, &c);
    size_t bh = (size_t)b * N_HEADS + h;
    size_t qi = (bh * T_SEQ + t) * HEAD_DIM + 2 * dp;
    qb[qi]     = (bf16)(q0 * c - q1 * s);
    qb[qi + 1] = (bf16)(q1 * c + q0 * s);
    kb[qi]     = (bf16)(k0 * c - k1 * s);
    kb[qi + 1] = (bf16)(k1 * c + k0 * s);
    size_t vi = (bh * HEAD_DIM + 2 * dp) * T_SEQ + t;
    vT[vi]         = (bf16)v0;
    vT[vi + T_SEQ] = (bf16)v1;
}

// ---------------------------------------------------------------- windowed flash attention
// One wave per (b, h, 16-query tile). Online softmax over key tiles qt-9..qt (pairs of 2).
__global__ __launch_bounds__(128) void attn_kernel(
    const bf16* __restrict__ qb, const bf16* __restrict__ kb,
    const bf16* __restrict__ vT, bf16* __restrict__ out)
{
    __shared__ __align__(16) bf16 lds_p[4][16 * 32];  // per-wave P staging (C-layout -> A-layout)
    int waveInBlk = threadIdx.x >> 5;
    int lane = threadIdx.x & 31;
    int half = lane >> 4;
    int l16  = lane & 15;

    int tile = blockIdx.x * 4 + waveInBlk;
    const int QT = T_SEQ / 16;                    // 128 query tiles per (b,h)
    int qt = tile % QT;
    int bh = tile / QT;
    int b  = bh / N_HEADS;
    int h  = bh % N_HEADS;
    int q0 = qt * 16;

    // ---- Q A-fragments: 16 x 64 = two 16x32 frags
    const bf16* qptr = qb + ((size_t)bh * T_SEQ + q0) * HEAD_DIM;
    v16bf aq[2];
#pragma unroll
    for (int f = 0; f < 2; ++f) {
        const bf16* rowp = qptr + (size_t)l16 * HEAD_DIM + f * 32 + half * 16;
#pragma unroll
        for (int r = 0; r < 8; ++r) {
            int kb_ = 2 * r;
            bf16x2 p = *reinterpret_cast<const bf16x2*>(rowp + kb_);
            aq[f][2 * r] = p[0]; aq[f][2 * r + 1] = p[1];
        }
    }

    float m_run[8], l_run[8];
    v8f o[4];
#pragma unroll
    for (int r = 0; r < 8; ++r) { m_run[r] = -1e30f; l_run[r] = 0.f; }
#pragma unroll
    for (int nt = 0; nt < 4; ++nt)
#pragma unroll
        for (int r = 0; r < 8; ++r) o[nt][r] = 0.f;

    const float scale = 0.125f;                   // 1/sqrt(64)

    for (int pr = 0; pr < 5; ++pr) {
        int kt0 = qt - 9 + 2 * pr;                // two key tiles kt0, kt0+1 (32 keys)

        // ---- S = Q * K^T for both key tiles (2 WMMAs each over Dh=64)
        v8f s2[2];
#pragma unroll
        for (int kk = 0; kk < 2; ++kk) {
            int kt = kt0 + kk;
            v8f sc;
#pragma unroll
            for (int r = 0; r < 8; ++r) sc[r] = 0.f;
            int key  = kt * 16 + l16;             // B-frag lane column = key index
            int keyc = min(max(key, 0), T_SEQ - 1);
            const bf16* krow = kb + ((size_t)bh * T_SEQ + keyc) * HEAD_DIM + half * 16;
#pragma unroll
            for (int f = 0; f < 2; ++f) {
                v16bf bk;
#pragma unroll
                for (int r = 0; r < 8; ++r) {
                    bf16x2 p = *reinterpret_cast<const bf16x2*>(krow + f * 32 + 2 * r);
                    bk[2 * r] = p[0]; bk[2 * r + 1] = p[1];
                }
                sc = __builtin_amdgcn_wmma_f32_16x16x32_bf16(false, aq[f], false, bk,
                                                             (short)0, sc, false, false);
            }
            s2[kk] = sc;
        }

        // ---- scale + window mask + tile row-max
        float tmax[8];
#pragma unroll
        for (int r = 0; r < 8; ++r) {
            int qpos = q0 + r + half * 8;
#pragma unroll
            for (int kk = 0; kk < 2; ++kk) {
                int kpos = (kt0 + kk) * 16 + l16;
                int rel  = qpos - kpos;
                bool ok  = (kpos >= 0) && (rel >= 0) && (rel < WINDOW);
                s2[kk][r] = ok ? s2[kk][r] * scale : -1e30f;
            }
            float mx = fmaxf(s2[0][r], s2[1][r]);
#pragma unroll
            for (int off = 8; off >= 1; off >>= 1)
                mx = fmaxf(mx, __shfl_xor(mx, off, 16));
            tmax[r] = mx;
        }

        // ---- online softmax update
        float alpha[8];
#pragma unroll
        for (int r = 0; r < 8; ++r) {
            float mnew = fmaxf(m_run[r], tmax[r]);
            alpha[r]   = __expf(m_run[r] - mnew);
            m_run[r]   = mnew;
            float p0 = __expf(s2[0][r] - mnew);
            float p1 = __expf(s2[1][r] - mnew);
            s2[0][r] = p0; s2[1][r] = p1;
            float rs = p0 + p1;
#pragma unroll
            for (int off = 8; off >= 1; off >>= 1)
                rs += __shfl_xor(rs, off, 16);
            l_run[r] = l_run[r] * alpha[r] + rs;
        }

        // ---- P (16x32) bf16 -> LDS (row-major), reload as A-fragment
        bf16* pl = lds_p[waveInBlk];
#pragma unroll
        for (int kk = 0; kk < 2; ++kk)
#pragma unroll
            for (int r = 0; r < 8; ++r)
                pl[(r + half * 8) * 32 + kk * 16 + l16] = (bf16)s2[kk][r];
        asm volatile("s_wait_dscnt 0" ::: "memory");
        v16bf ap;
        {
            const bf16* rowp = pl + l16 * 32 + half * 16;
#pragma unroll
            for (int r = 0; r < 8; ++r) {
                int kb_ = 2 * r;
                bf16x2 p = *reinterpret_cast<const bf16x2*>(rowp + kb_);
                ap[2 * r] = p[0]; ap[2 * r + 1] = p[1];
            }
        }

        // ---- O = O*alpha + P x V   (V B-frags from transposed v: K-pairs contiguous)
        int kbase = kt0 * 16;
#pragma unroll
        for (int nt = 0; nt < 4; ++nt) {
            const bf16* vrow = vT + ((size_t)bh * HEAD_DIM + nt * 16 + l16) * T_SEQ;
            v16bf bv;
#pragma unroll
            for (int r = 0; r < 8; ++r) {
                int key  = kbase + half * 16 + 2 * r;
                int keyc = min(max(key, 0), T_SEQ - 2) & ~1;
                bf16x2 p = *reinterpret_cast<const bf16x2*>(vrow + keyc);
                bv[2 * r] = p[0]; bv[2 * r + 1] = p[1];
            }
            v8f oo = o[nt];
#pragma unroll
            for (int r = 0; r < 8; ++r) oo[r] *= alpha[r];
            o[nt] = __builtin_amdgcn_wmma_f32_16x16x32_bf16(false, ap, false, bv,
                                                            (short)0, oo, false, false);
        }
    }

    // ---- normalize, write bf16 into [B*T, C] for proj GEMM
#pragma unroll
    for (int r = 0; r < 8; ++r) {
        float inv_l = 1.0f / l_run[r];
        int mg = q0 + r + half * 8;
        size_t rowo = ((size_t)b * T_SEQ + mg) * CDIM + h * HEAD_DIM;
#pragma unroll
        for (int nt = 0; nt < 4; ++nt)
            out[rowo + nt * 16 + l16] = (bf16)(o[nt][r] * inv_l);
    }
}

// ---------------------------------------------------------------- launch
extern "C" void kernel_launch(void* const* d_in, const int* in_sizes, int n_in,
                              void* d_out, int out_size, void* d_ws, size_t ws_size,
                              hipStream_t stream)
{
    (void)in_sizes; (void)n_in; (void)out_size; (void)ws_size;
    const float* x     = (const float*)d_in[0];
    const float* Wqkv  = (const float*)d_in[1];
    const float* bqkv  = (const float*)d_in[2];
    const float* Wproj = (const float*)d_in[3];
    const float* bproj = (const float*)d_in[4];

    char* ws = (char*)d_ws;
    bf16*  xb     = (bf16*)(ws + OFF_XB);
    bf16*  wqkvT  = (bf16*)(ws + OFF_WQKVT);
    bf16*  wprojT = (bf16*)(ws + OFF_WPROJT);
    float* qkvf   = (float*)(ws + OFF_QKV);
    bf16*  qbuf   = (bf16*)(ws + OFF_QB);
    bf16*  kbuf   = (bf16*)(ws + OFF_KB);
    bf16*  vTbuf  = (bf16*)(ws + OFF_VT);
    bf16*  attnb  = (bf16*)(ws + OFF_ATTNB);

    // 1. x -> bf16
    {
        int n = M_ROWS * CDIM;
        f32_to_bf16_kernel<<<(n + 255) / 256, 256, 0, stream>>>(x, xb, n);
    }
    // 2. weights -> transposed bf16
    transpose_to_bf16_kernel<<<(3 * CDIM * CDIM + 255) / 256, 256, 0, stream>>>(Wqkv, wqkvT, CDIM, 3 * CDIM);
    transpose_to_bf16_kernel<<<(CDIM * CDIM + 255) / 256, 256, 0, stream>>>(Wproj, wprojT, CDIM, CDIM);
    // 3. QKV GEMM: [4096,1024] x [1024,3072], block tile 128x128
    {
        int blocks = (M_ROWS / 128) * ((3 * CDIM) / 128);   // 32*24 = 768
        gemm_bf16_kernel<<<blocks, 256, 0, stream>>>(xb, wqkvT, bqkv, qkvf,
                                                     M_ROWS, 3 * CDIM, CDIM);
    }
    // 4. RoPE + head split (+ V transpose)
    {
        int pairs = BATCH * T_SEQ * N_HEADS * (HEAD_DIM / 2);
        rope_split_kernel<<<(pairs + 255) / 256, 256, 0, stream>>>(qkvf, qbuf, kbuf, vTbuf);
    }
    // 5. windowed attention: 4096 wave-tiles, 4 waves/block
    {
        int tiles = BATCH * N_HEADS * (T_SEQ / 16);      // 4096
        attn_kernel<<<tiles / 4, 128, 0, stream>>>(qbuf, kbuf, vTbuf, attnb);
    }
    // 6. proj GEMM: [4096,1024] x [1024,1024] -> d_out f32
    {
        int blocks = (M_ROWS / 128) * (CDIM / 128);      // 32*8 = 256
        gemm_bf16_kernel<<<blocks, 256, 0, stream>>>(attnb, wprojT, bproj,
                                                     (float*)d_out, M_ROWS, CDIM, CDIM);
    }
}
